// LinearAttention_39384850104920
// MI455X (gfx1250) — matrix operand order repaired and checked by reading, not compiled, not run on hardware
//
#include <hip/hip_runtime.h>
#include <hip/hip_bf16.h>

// ---------------------------------------------------------------------------
// Linear attention (elu+1 feature map) for MI455X / gfx1250.
// All GEMMs via v_wmma_f32_16x16x32_bf16 (wave32, 16x16 tiles, f32 accum).
// Weight/x pre-converted to bf16; LDS tiles staged with
// global_load_async_to_lds_b128 (ASYNCcnt) when available, double-buffered.
// B=4, T=4096, d=1024, H=16, D=64.
// ---------------------------------------------------------------------------

typedef __attribute__((ext_vector_type(16))) __bf16 v16bf;
typedef __attribute__((ext_vector_type(8)))  float  v8f;
typedef __attribute__((ext_vector_type(4)))  int    v4i;

#define B_   4
#define T_   4096
#define H_   16
#define D_   64
#define DM   1024
#define BT_  (B_ * T_)
#define EPS_ 1e-6f

// ---- CDNA5 async global->LDS path (guarded; sync fallback keeps it green) --
#if defined(__has_builtin)
#  if __has_builtin(__builtin_amdgcn_global_load_async_to_lds_b128)
#    define HAVE_ASYNC_B128 1
#  endif
#  if __has_builtin(__builtin_amdgcn_s_wait_asynccnt)
#    define HAVE_WAIT_ASYNC 1
#  endif
#endif
#ifndef HAVE_ASYNC_B128
#  define HAVE_ASYNC_B128 0
#endif
#ifndef HAVE_WAIT_ASYNC
#  define HAVE_WAIT_ASYNC 0
#endif

typedef __attribute__((address_space(1))) v4i* gv4i_p;   // global int4*
typedef __attribute__((address_space(3))) v4i* lv4i_p;   // LDS int4*

__device__ __forceinline__ void async_copy16(const __bf16* g, __bf16* l) {
#if HAVE_ASYNC_B128
  __builtin_amdgcn_global_load_async_to_lds_b128(
      (gv4i_p)(g), (lv4i_p)(l), /*offset=*/0, /*cpol=*/0);
#else
  *(uint4*)l = *(const uint4*)g;
#endif
}

__device__ __forceinline__ void async_wait0() {
#if HAVE_ASYNC_B128 && HAVE_WAIT_ASYNC
  __builtin_amdgcn_s_wait_asynccnt(0);
#endif
}

__device__ __forceinline__ v8f wmma_bf16(v16bf a, v16bf b, v8f c) {
  // (neg_a, A, neg_b, B, c_mod, C, reuse_a, reuse_b)
  return __builtin_amdgcn_wmma_f32_16x16x32_bf16(false, a, false, b, (short)0, c,
                                                 false, false);
}

// elu(x)+1 : x>0 -> x+1 ; x<=0 -> exp(x)
__device__ __forceinline__ float phi_f(float x) {
  return x > 0.f ? x + 1.f : __expf(x);
}

// ===========================================================================
// Kernel 0: f32 -> bf16 conversion (x and weights), 8 elems/thread, 16B store
// ===========================================================================
__global__ __launch_bounds__(256) void cvt_bf16_kernel(
    const float* __restrict__ src, __bf16* __restrict__ dst, int n8)
{
  const int i = blockIdx.x * 256 + threadIdx.x;
  if (i >= n8) return;
  const float4* s = (const float4*)src + (size_t)i * 2;
  const float4 lo = s[0], hi = s[1];
  union { __bf16 h[8]; uint4 u; } t;
  t.h[0] = (__bf16)lo.x; t.h[1] = (__bf16)lo.y;
  t.h[2] = (__bf16)lo.z; t.h[3] = (__bf16)lo.w;
  t.h[4] = (__bf16)hi.x; t.h[5] = (__bf16)hi.y;
  t.h[6] = (__bf16)hi.z; t.h[7] = (__bf16)hi.w;
  *(uint4*)(dst + (size_t)i * 8) = t.u;
}

// ===========================================================================
// Kernel 1: fused Q/K/V projections.  Y = x * W^T  (M=BT, N=DM, K=DM), bf16.
//   mode 0: Q -> phi,      store Qp [B,H,T,D]
//   mode 1: K -> phi*mask, store Kt [B,H,D,T] (transposed)
//   mode 2: V -> mask,     store Vt [B,H,D,T]
// Block = 128 thr (4 waves); wave = 16(M) x 128(N); block = 64 x 128.
// W tile (32K x 128N bf16, 8KB) double-buffered in LDS via async copies.
// ===========================================================================
#define NT_ 8
__global__ __launch_bounds__(128) void qkv_proj_kernel(
    const __bf16* __restrict__ xb, const unsigned char* __restrict__ mask,
    const __bf16* __restrict__ Wqb, const __bf16* __restrict__ Wkb,
    const __bf16* __restrict__ Wvb,
    __bf16* __restrict__ Qp, __bf16* __restrict__ Kt, __bf16* __restrict__ Vt)
{
  __shared__ __bf16 lds_w[2][128 * 32];        // [buf][n][k], 16 KB

  const int mode = blockIdx.z;
  const __bf16* W = (mode == 0) ? Wqb : ((mode == 1) ? Wkb : Wvb);
  const int wave = threadIdx.x >> 5;
  const int lane = threadIdx.x & 31;
  const int hi   = lane >> 4;
  const int lm   = lane & 15;
  const int row0 = blockIdx.x * 64 + wave * 16;   // M base (in BT)
  const int n0   = blockIdx.y * 128;              // N base

  // staging assignment: thread copies 4 x 16B; rows sn, sn+32, sn+64, sn+96
  const int sn   = threadIdx.x >> 2;              // 0..31
  const int sseg = (threadIdx.x & 3) * 8;         // bf16 offset within row

  v8f acc[NT_] = {};

  // prologue: stage tile kt=0 into buffer 0
#pragma unroll
  for (int r = 0; r < 4; ++r) {
    const int row = r * 32 + sn;
    async_copy16(W + (size_t)(n0 + row) * DM + sseg,
                 &lds_w[0][row * 32 + sseg]);
  }

  const __bf16* arow = xb + (size_t)(row0 + lm) * DM;

  for (int kt = 0; kt < DM; kt += 32) {
    const int cur = (kt >> 5) & 1;
    async_wait0();
    __syncthreads();
    if (kt + 32 < DM) {
#pragma unroll
      for (int r = 0; r < 4; ++r) {
        const int row = r * 32 + sn;
        async_copy16(W + (size_t)(n0 + row) * DM + kt + 32 + sseg,
                     &lds_w[cur ^ 1][row * 32 + sseg]);
      }
    }

    // A fragment (bf16 direct, two contiguous 16B loads)
    v16bf a;
    const __bf16* as = arow + kt + hi * 8;
#pragma unroll
    for (int j = 0; j < 8; ++j) a[j] = as[j];
#pragma unroll
    for (int j = 0; j < 8; ++j) a[8 + j] = as[16 + j];

#pragma unroll
    for (int nt = 0; nt < NT_; ++nt) {
      v16bf b;
      const __bf16* bs = &lds_w[cur][(nt * 16 + lm) * 32 + hi * 8];
#pragma unroll
      for (int j = 0; j < 16; ++j) b[j] = bs[(j < 8) ? j : j + 8];
      acc[nt] = wmma_bf16(a, b, acc[nt]);
    }
  }

  // epilogue: C layout row m = vgpr + 8*hi, col n = lm
#pragma unroll
  for (int nt = 0; nt < NT_; ++nt) {
    const int e   = n0 + nt * 16 + lm;
    const int h   = e >> 6;
    const int dch = e & 63;
#pragma unroll
    for (int v = 0; v < 8; ++v) {
      const int m  = v + 8 * hi;
      const int r  = row0 + m;
      const int bb = r >> 12;                   // r / T_
      const int t  = r & (T_ - 1);              // r % T_
      const float c = acc[nt][v];
      if (mode == 0) {
        Qp[(((size_t)(bb * H_ + h)) * T_ + t) * D_ + dch] = (__bf16)phi_f(c);
      } else {
        const float mk = mask[bb * T_ + t] ? 1.f : 0.f;
        const float val = (mode == 1) ? phi_f(c) * mk : c * mk;
        __bf16* dst = (mode == 1) ? Kt : Vt;
        dst[(((size_t)(bb * H_ + h)) * D_ + dch) * T_ + t] = (__bf16)val;
      }
    }
  }
}

// ===========================================================================
// Kernel 2: per (b,h): KV = K^T V (64x64, reduce over T) and K_sum (64).
// 4 waves; wave owns 16 d-rows.  KV stored transposed [e][d] bf16.
// ===========================================================================
__global__ __launch_bounds__(128) void kv_kernel(
    const __bf16* __restrict__ Kt, const __bf16* __restrict__ Vt,
    __bf16* __restrict__ KVt, float* __restrict__ Ksum)
{
  const int bh   = blockIdx.x;
  const int wave = threadIdx.x >> 5;
  const int lane = threadIdx.x & 31;
  const int hi   = lane >> 4;
  const int lm   = lane & 15;
  const int drow = wave * 16 + lm;

  v8f acc[4] = {};
  float ks = 0.f;

  const __bf16* Krow = Kt + ((size_t)bh * D_ + drow) * T_;

  for (int t0 = 0; t0 < T_; t0 += 32) {
    __builtin_prefetch(Krow + t0 + 1024, 0, 1);      // global_prefetch
    v16bf a;
#pragma unroll
    for (int j = 0; j < 8; ++j) a[j] = Krow[t0 + hi * 8 + j];
#pragma unroll
    for (int j = 0; j < 8; ++j) a[8 + j] = Krow[t0 + 16 + hi * 8 + j];
#pragma unroll
    for (int j = 0; j < 16; ++j) ks += (float)a[j];  // K_sum partial

#pragma unroll
    for (int nt = 0; nt < 4; ++nt) {
      const int e = nt * 16 + lm;
      const __bf16* Vrow = Vt + ((size_t)bh * D_ + e) * T_ + t0 + hi * 8;
      v16bf b;
#pragma unroll
      for (int j = 0; j < 16; ++j) b[j] = Vrow[(j < 8) ? j : j + 8];
      acc[nt] = wmma_bf16(a, b, acc[nt]);
    }
  }

  // KV^T store: KVt[bh][e][d], 8 consecutive d per lane
#pragma unroll
  for (int nt = 0; nt < 4; ++nt) {
    const int e = nt * 16 + lm;
#pragma unroll
    for (int v = 0; v < 8; ++v) {
      const int m = v + 8 * hi;
      KVt[(size_t)bh * D_ * D_ + (size_t)e * D_ + wave * 16 + m] =
          (__bf16)acc[nt][v];
    }
  }

  ks += __shfl_xor(ks, 16, 32);   // merge half-wave K partials (wave32)
  if (hi == 0) Ksum[bh * D_ + drow] = ks;
}

// ===========================================================================
// Kernel 3: context = Q * KV, normalized by Z = Q.K_sum + eps.
// Grid (T/64, B*H); wave = 16 t-rows x 64 e-cols.  KV^T staged via async.
// Writes ctx as [B*T, 1024] bf16 (GEMM-ready for output projection).
// ===========================================================================
__global__ __launch_bounds__(128) void ctx_kernel(
    const __bf16* __restrict__ Qp, const __bf16* __restrict__ KVt,
    const float* __restrict__ Ksum, __bf16* __restrict__ ctx)
{
  __shared__ __bf16 lds_kv[D_ * D_];           // [e][d], 8 KB
  __shared__ float  lds_ks[D_];
  __shared__ float  lds_z[4][16];

  const int bh   = blockIdx.y;
  const int wave = threadIdx.x >> 5;
  const int lane = threadIdx.x & 31;
  const int hi   = lane >> 4;
  const int lm   = lane & 15;
  const int trow = blockIdx.x * 64 + wave * 16 + lm;

  // stage KV^T via async copies (4 x 16B per thread) + K_sum
#pragma unroll
  for (int c = 0; c < 4; ++c) {
    const int idx16 = threadIdx.x + c * 128;   // 16B chunk index, 0..511
    async_copy16(KVt + (size_t)bh * D_ * D_ + idx16 * 8, &lds_kv[idx16 * 8]);
  }
  if (threadIdx.x < D_) lds_ks[threadIdx.x] = Ksum[bh * D_ + threadIdx.x];
  async_wait0();
  __syncthreads();

  v8f acc[4] = {};
  float z = 0.f;
  const __bf16* Qrow = Qp + ((size_t)bh * T_ + trow) * D_;

#pragma unroll
  for (int kstep = 0; kstep < 2; ++kstep) {    // K dim = 64 = 2 x 32
    const int k0 = kstep * 32;
    v16bf a;
#pragma unroll
    for (int j = 0; j < 8; ++j) {
      const int k = k0 + hi * 8 + j;
      a[j] = Qrow[k];
      z += (float)a[j] * lds_ks[k];
    }
#pragma unroll
    for (int j = 0; j < 8; ++j) {
      const int k = k0 + 16 + hi * 8 + j;
      a[8 + j] = Qrow[k];
      z += (float)a[8 + j] * lds_ks[k];
    }
#pragma unroll
    for (int nt = 0; nt < 4; ++nt) {
      v16bf b;
      const __bf16* bs = &lds_kv[(nt * 16 + lm) * D_ + k0 + hi * 8];
#pragma unroll
      for (int j = 0; j < 16; ++j) b[j] = bs[(j < 8) ? j : j + 8];
      acc[nt] = wmma_bf16(a, b, acc[nt]);
    }
  }

  z += __shfl_xor(z, 16, 32);                  // full-row dot(Q, K_sum)
  if (hi == 0) lds_z[wave][lm] = z + EPS_;
  __syncthreads();

  const int b = bh / H_;
  const int h = bh % H_;
#pragma unroll
  for (int nt = 0; nt < 4; ++nt) {
    const int e = nt * 16 + lm;
#pragma unroll
    for (int v = 0; v < 8; ++v) {
      const int m  = v + 8 * hi;
      const int t  = blockIdx.x * 64 + wave * 16 + m;
      const float zz = lds_z[wave][m];
      ctx[((size_t)(b * T_ + t)) * DM + h * D_ + e] = (__bf16)(acc[nt][v] / zz);
    }
  }
}

// ===========================================================================
// Kernel 4: out = ctx * Wo^T + bo  (M=BT, N=DM, K=DM), f32 output.
// Same async double-buffered tiling as kernel 1.
// ===========================================================================
__global__ __launch_bounds__(128) void out_proj_kernel(
    const __bf16* __restrict__ ctx, const __bf16* __restrict__ Wob,
    const float* __restrict__ bo, float* __restrict__ out)
{
  __shared__ __bf16 lds_w[2][128 * 32];

  const int wave = threadIdx.x >> 5;
  const int lane = threadIdx.x & 31;
  const int hi   = lane >> 4;
  const int lm   = lane & 15;
  const int row0 = blockIdx.x * 64 + wave * 16;
  const int n0   = blockIdx.y * 128;

  const int sn   = threadIdx.x >> 2;
  const int sseg = (threadIdx.x & 3) * 8;

  v8f acc[NT_] = {};

#pragma unroll
  for (int r = 0; r < 4; ++r) {
    const int row = r * 32 + sn;
    async_copy16(Wob + (size_t)(n0 + row) * DM + sseg,
                 &lds_w[0][row * 32 + sseg]);
  }

  const __bf16* arow = ctx + (size_t)(row0 + lm) * DM;

  for (int kt = 0; kt < DM; kt += 32) {
    const int cur = (kt >> 5) & 1;
    async_wait0();
    __syncthreads();
    if (kt + 32 < DM) {
#pragma unroll
      for (int r = 0; r < 4; ++r) {
        const int row = r * 32 + sn;
        async_copy16(Wob + (size_t)(n0 + row) * DM + kt + 32 + sseg,
                     &lds_w[cur ^ 1][row * 32 + sseg]);
      }
    }

    v16bf a;
    const __bf16* as = arow + kt + hi * 8;
#pragma unroll
    for (int j = 0; j < 8; ++j) a[j] = as[j];
#pragma unroll
    for (int j = 0; j < 8; ++j) a[8 + j] = as[16 + j];

#pragma unroll
    for (int nt = 0; nt < NT_; ++nt) {
      v16bf b;
      const __bf16* bs = &lds_w[cur][(nt * 16 + lm) * 32 + hi * 8];
#pragma unroll
      for (int j = 0; j < 16; ++j) b[j] = bs[(j < 8) ? j : j + 8];
      acc[nt] = wmma_bf16(a, b, acc[nt]);
    }
  }

#pragma unroll
  for (int nt = 0; nt < NT_; ++nt) {
    const int e = n0 + nt * 16 + lm;
    const float bias = bo[e];
#pragma unroll
    for (int v = 0; v < 8; ++v) {
      const int m = v + 8 * hi;
      out[(size_t)(row0 + m) * DM + e] = acc[nt][v] + bias;
    }
  }
}

// ===========================================================================
extern "C" void kernel_launch(void* const* d_in, const int* in_sizes, int n_in,
                              void* d_out, int out_size, void* d_ws,
                              size_t ws_size, hipStream_t stream) {
  const float*         x    = (const float*)d_in[0];
  const unsigned char* mask = (const unsigned char*)d_in[1];  // jnp bool = u8
  const float*         Wq   = (const float*)d_in[2];
  const float*         Wk   = (const float*)d_in[3];
  const float*         Wv   = (const float*)d_in[4];
  const float*         Wo   = (const float*)d_in[5];
  const float*         bo   = (const float*)d_in[6];

  char* ws = (char*)d_ws;
  size_t off = 0;
  __bf16* xb   = (__bf16*)(ws + off); off += (size_t)BT_ * DM * 2;          // 32MB
  __bf16* Wqb  = (__bf16*)(ws + off); off += (size_t)DM * DM * 2;           // 2MB
  __bf16* Wkb  = (__bf16*)(ws + off); off += (size_t)DM * DM * 2;
  __bf16* Wvb  = (__bf16*)(ws + off); off += (size_t)DM * DM * 2;
  __bf16* Wob  = (__bf16*)(ws + off); off += (size_t)DM * DM * 2;
  __bf16* Qp   = (__bf16*)(ws + off); off += (size_t)B_ * H_ * T_ * D_ * 2; // 32MB
  __bf16* Kt   = (__bf16*)(ws + off); off += (size_t)B_ * H_ * T_ * D_ * 2; // 32MB
  __bf16* Vt   = (__bf16*)(ws + off); off += (size_t)B_ * H_ * T_ * D_ * 2; // 32MB
  __bf16* ctx  = (__bf16*)(ws + off); off += (size_t)BT_ * DM * 2;          // 32MB
  __bf16* KVt  = (__bf16*)(ws + off); off += (size_t)B_ * H_ * D_ * D_ * 2;
  float*  Ksum = (float*)(ws + off);  off += (size_t)B_ * H_ * D_ * 4;

  // ---- pre-convert to bf16 ----
  const int n8x = BT_ * DM / 8;                 // 2M
  const int n8w = DM * DM / 8;                  // 128K
  cvt_bf16_kernel<<<dim3((n8x + 255) / 256), 256, 0, stream>>>(x, xb, n8x);
  cvt_bf16_kernel<<<dim3((n8w + 255) / 256), 256, 0, stream>>>(Wq, Wqb, n8w);
  cvt_bf16_kernel<<<dim3((n8w + 255) / 256), 256, 0, stream>>>(Wk, Wkb, n8w);
  cvt_bf16_kernel<<<dim3((n8w + 255) / 256), 256, 0, stream>>>(Wv, Wvb, n8w);
  cvt_bf16_kernel<<<dim3((n8w + 255) / 256), 256, 0, stream>>>(Wo, Wob, n8w);

  // ---- pipeline ----
  qkv_proj_kernel<<<dim3(BT_ / 64, DM / 128, 3), 128, 0, stream>>>(
      xb, mask, Wqb, Wkb, Wvb, Qp, Kt, Vt);
  kv_kernel<<<dim3(B_ * H_), 128, 0, stream>>>(Kt, Vt, KVt, Ksum);
  ctx_kernel<<<dim3(T_ / 64, B_ * H_), 128, 0, stream>>>(Qp, KVt, Ksum, ctx);
  out_proj_kernel<<<dim3(BT_ / 64, DM / 128), 128, 0, stream>>>(
      ctx, Wob, bo, (float*)d_out);
}
